// GCNEncoder_64785286693470
// MI455X (gfx1250) — matrix operand order repaired and checked by reading, not compiled, not run on hardware
//
#include <hip/hip_runtime.h>
#include <hip/hip_bf16.h>

// ---------- types ----------
typedef __attribute__((ext_vector_type(16))) __bf16      v16bf;
typedef __attribute__((ext_vector_type(8)))  float       v8f;
typedef __attribute__((ext_vector_type(4)))  unsigned int u32x4;

#define D_FEAT 512

// ---------- small utility kernels ----------
__global__ void zero_f32(float* p, long long n) {
    long long i = (long long)blockIdx.x * blockDim.x + threadIdx.x;
    if (i < n) p[i] = 0.0f;
}

__global__ void cvt_f32_bf16(const float* __restrict__ in, __bf16* __restrict__ out,
                             long long n) {
    long long i = (long long)blockIdx.x * blockDim.x + threadIdx.x;
    if (i < n) out[i] = (__bf16)in[i];
}

// z -> hb = bf16(relu(z)), zb = bf16(z)
__global__ void relu_cvt(const float* __restrict__ z, __bf16* __restrict__ hb,
                         __bf16* __restrict__ zb, long long n) {
    long long i = (long long)blockIdx.x * blockDim.x + threadIdx.x;
    if (i < n) {
        float v = z[i];
        zb[i] = (__bf16)v;
        hb[i] = (__bf16)(v > 0.0f ? v : 0.0f);
    }
}

// degree with self loops: deg[dst[e]] += 1 for all edges, deg[i] += 1 for loops
__global__ void degree_k(const int* __restrict__ dst, float* __restrict__ deg,
                         int E, int Nn) {
    int t = blockIdx.x * blockDim.x + threadIdx.x;
    if (t < E) {
        atomicAdd(&deg[dst[t]], 1.0f);
    } else if (t < E + Nn) {
        atomicAdd(&deg[t - E], 1.0f);
    }
}

__global__ void rsqrt_k(const float* __restrict__ deg, float* __restrict__ dis, int Nn) {
    int i = blockIdx.x * blockDim.x + threadIdx.x;
    if (i < Nn) dis[i] = rsqrtf(fmaxf(deg[i], 1.0f));
}

// out[dst] += h[src] * (dis[src]*dis[dst]); one block per edge (incl. self loops),
// 128 threads x float4 = 512 features. Scatter-add via global_atomic_add_f32 (L2).
__global__ __launch_bounds__(128)
void aggregate_k(const float* __restrict__ h, const int* __restrict__ src,
                 const int* __restrict__ dst, const float* __restrict__ dis,
                 float* __restrict__ out, int E) {
    const int e = blockIdx.x;
    int s, d; float n;
    if (e < E) {
        s = src[e]; d = dst[e];
        n = dis[s] * dis[d];
    } else {
        s = d = e - E;
        float t = dis[s];
        n = t * t;
    }
    const int c = threadIdx.x;                 // 0..127
    const float4 v = ((const float4*)(h + (long long)s * D_FEAT))[c];
    float* o = out + (long long)d * D_FEAT + c * 4;
    atomicAdd(o + 0, v.x * n);
    atomicAdd(o + 1, v.y * n);
    atomicAdd(o + 2, v.z * n);
    atomicAdd(o + 3, v.w * n);
}

// ---------- bf16 WMMA GEMM: C[M,Nc] = A[M,K] @ W[K,Nc] + bias ----------
// Block = 256 threads (8 waves). Block tile: 64(M) x 128(N); each wave owns a
// 16-column strip and 4 stacked 16x16 M-subtiles. A panel is staged into
// double-buffered LDS via the CDNA5 async copy engine
// (global_load_async_to_lds_b128, tracked by ASYNCcnt), overlapping the copy
// of tile t+1 with the WMMAs of tile t.
template <bool RELU, bool WF32, bool WBF16>
__global__ __launch_bounds__(256)
void gemm_bf16(const __bf16* __restrict__ A, const __bf16* __restrict__ W,
               const float* __restrict__ bias, float* __restrict__ Cf,
               __bf16* __restrict__ Cb, int M, int K, int Nc) {
    __shared__ __align__(16) __bf16 ldsA[2][64 * 32];   // 2 x 4KB ping-pong

    const int lane = threadIdx.x & 31;
    const int wid  = threadIdx.x >> 5;
    const int half = lane >> 4;            // 0: lanes 0-15, 1: lanes 16-31
    const int l15  = lane & 15;

    const long long rowBase = (long long)blockIdx.x * 64;
    const int       colBase = blockIdx.y * 128 + wid * 16;

    v8f acc[4];
    for (int i = 0; i < 4; ++i)
        for (int r = 0; r < 8; ++r) acc[i][r] = 0.0f;

    // cooperative A-stage mapping: 256 threads x 16B covers 64x32 bf16
    const int sRow = threadIdx.x >> 2;        // 0..63
    const int sOff = (threadIdx.x & 3) * 8;   // bf16 offset within row of 32
    const long long gRowS = rowBase + sRow;
    // rows >= M skip the copy and leave garbage in LDS; harmless because
    // WMMA output row m depends only on A row m, and rows >= M are not stored.
    const bool stageOk = (gRowS < M);
    const __bf16* pAsrc = A + gRowS * (long long)K + sOff;
    const unsigned ldsDst0 = (unsigned)(uintptr_t)(&ldsA[0][sRow * 32 + sOff]);
    const unsigned ldsDst1 = (unsigned)(uintptr_t)(&ldsA[1][sRow * 32 + sOff]);

    const int nIter = K >> 5;   // K / 32

    // async-stage tile 0 -> buffer 0
    if (stageOk) {
        asm volatile("global_load_async_to_lds_b128 %0, %1, off"
                     :: "v"(ldsDst0), "v"((unsigned long long)(uintptr_t)pAsrc)
                     : "memory");
    }

    for (int t = 0; t < nIter; ++t) {
        // issue async copy of tile t+1 into the other buffer (its previous
        // readers finished at the end-of-iteration barrier of iter t-1)
        if (t + 1 < nIter) {
            if (stageOk) {
                const __bf16* p = pAsrc + (long long)(t + 1) * 32;
                asm volatile("global_load_async_to_lds_b128 %0, %1, off"
                             :: "v"((t & 1) ? ldsDst0 : ldsDst1),
                                "v"((unsigned long long)(uintptr_t)p)
                             : "memory");
            }
            // async loads complete in order: <=1 outstanding => tile t resident
            asm volatile("s_wait_asynccnt 0x1" ::: "memory");
        } else {
            asm volatile("s_wait_asynccnt 0x0" ::: "memory");
        }
        __syncthreads();

        // B fragment: lane holds col n = colBase + l15, element s at k = 32t + 8*half + (s&7) + 16*(s>>3)
        union { v16bf v; u32x4 q[2]; } bf;
        const int krow = t * 32 + 8 * half;
        const __bf16* pB = W + (long long)krow * Nc + colBase + l15;
        for (int s = 0; s < 16; ++s)
            bf.v[s] = pB[(long long)((s & 7) + (s >> 3) * 16) * Nc];
        if (t + 1 < nIter) __builtin_prefetch(pB + 32 * (long long)Nc, 0, 1);

        const __bf16* lbase = ldsA[t & 1];

        // 4 M-subtiles reuse this B fragment
        for (int ms = 0; ms < 4; ++ms) {
            union { v16bf v; u32x4 q[2]; } af;
            const int ml = ms * 16 + l15;
            // matches 16-bit A layout: VGPR0-3 hold K = half*8 + 0..7,
            //                           VGPR4-7 hold K = 16 + half*8 + 0..7
            af.q[0] = *(const u32x4*)(lbase + ml * 32 + half * 8);
            af.q[1] = *(const u32x4*)(lbase + ml * 32 + 16 + half * 8);
            acc[ms] = __builtin_amdgcn_wmma_f32_16x16x32_bf16(
                false, af.v, false, bf.v, (short)0, acc[ms], false, false);
        }
        __syncthreads();   // all readers done before buffer is overwritten
    }

    // epilogue: C/D layout -> element r: M = r + 8*half, N = l15
    const int col = colBase + l15;
    const float bv = bias[col];
    for (int ms = 0; ms < 4; ++ms) {
        for (int r = 0; r < 8; ++r) {
            const long long row = rowBase + ms * 16 + r + 8 * half;
            if (row < M) {
                float v = acc[ms][r] + bv;
                if constexpr (RELU) v = v > 0.0f ? v : 0.0f;
                if constexpr (WF32)  Cf[row * (long long)Nc + col] = v;
                if constexpr (WBF16) Cb[row * (long long)Nc + col] = (__bf16)v;
            }
        }
    }
}

// ---------- driver ----------
extern "C" void kernel_launch(void* const* d_in, const int* in_sizes, int n_in,
                              void* d_out, int out_size, void* d_ws, size_t ws_size,
                              hipStream_t stream) {
    const float* x   = (const float*)d_in[0];
    const int*   ei  = (const int*)  d_in[1];
    const float* W1  = (const float*)d_in[2];
    const float* b1  = (const float*)d_in[3];
    const float* W2  = (const float*)d_in[4];
    const float* b2  = (const float*)d_in[5];
    const float* Wp1 = (const float*)d_in[6];
    const float* bp1 = (const float*)d_in[7];
    const float* Wp2 = (const float*)d_in[8];
    const float* bp2 = (const float*)d_in[9];

    const int D  = D_FEAT;
    const int Nn = in_sizes[0] / D;
    const int E  = in_sizes[1] / 2;
    const int* src = ei;
    const int* dst = ei + E;
    const long long nd = (long long)Nn * D;

    float* out  = (float*)d_out;          // [N, D]
    float* z    = out + nd;               // [N, D]
    float* proj = z + nd;                 // [N, D]

    // workspace carve-up
    char* w = (char*)d_ws;
    auto carve = [&](size_t bytes) -> void* {
        void* p = (void*)w;
        w += (bytes + 255) & ~(size_t)255;
        return p;
    };
    float*  deg  = (float*) carve((size_t)Nn * 4);
    float*  dis  = (float*) carve((size_t)Nn * 4);
    float*  fbuf = (float*) carve((size_t)nd * 4);   // h1 / h2 scratch
    __bf16* xb   = (__bf16*)carve((size_t)nd * 2);
    __bf16* hb   = (__bf16*)carve((size_t)nd * 2);
    __bf16* zb   = (__bf16*)carve((size_t)nd * 2);
    __bf16* W1b  = (__bf16*)carve((size_t)D * D * 2);
    __bf16* W2b  = (__bf16*)carve((size_t)D * D * 2);
    __bf16* Wp1b = (__bf16*)carve((size_t)D * D * 2);
    __bf16* Wp2b = (__bf16*)carve((size_t)D * D * 2);
    __bf16* p1b  = xb;  // xb dead after GEMM1; reuse for projection hidden

    const int T = 256;
    auto blks = [&](long long n) { return (unsigned)((n + T - 1) / T); };

    // init + conversions
    zero_f32<<<blks(Nn), T, 0, stream>>>(deg, Nn);
    zero_f32<<<blks(2 * nd), T, 0, stream>>>(out, 2 * nd);  // out + z accumulators
    cvt_f32_bf16<<<blks(nd), T, 0, stream>>>(x, xb, nd);
    cvt_f32_bf16<<<blks((long long)D * D), T, 0, stream>>>(W1,  W1b,  (long long)D * D);
    cvt_f32_bf16<<<blks((long long)D * D), T, 0, stream>>>(W2,  W2b,  (long long)D * D);
    cvt_f32_bf16<<<blks((long long)D * D), T, 0, stream>>>(Wp1, Wp1b, (long long)D * D);
    cvt_f32_bf16<<<blks((long long)D * D), T, 0, stream>>>(Wp2, Wp2b, (long long)D * D);

    // symmetric normalization with self loops
    degree_k<<<blks((long long)E + Nn), T, 0, stream>>>(dst, deg, E, Nn);
    rsqrt_k<<<blks(Nn), T, 0, stream>>>(deg, dis, Nn);

    dim3 gg((Nn + 63) / 64, D / 128);

    // layer 1: h1 = x@W1 + b1 ; z = scatter(h1)
    gemm_bf16<false, true, false><<<gg, 256, 0, stream>>>(xb, W1b, b1, fbuf, ((__bf16*)0), Nn, D, D);
    aggregate_k<<<E + Nn, 128, 0, stream>>>(fbuf, src, dst, dis, z, E);

    // h = relu(z) (bf16), zb = bf16(z)
    relu_cvt<<<blks(nd), T, 0, stream>>>(z, hb, zb, nd);

    // layer 2: h2 = h@W2 + b2 ; out = scatter(h2)
    gemm_bf16<false, true, false><<<gg, 256, 0, stream>>>(hb, W2b, b2, fbuf, ((__bf16*)0), Nn, D, D);
    aggregate_k<<<E + Nn, 128, 0, stream>>>(fbuf, src, dst, dis, out, E);

    // projection head: proj = relu(z@Wp1+bp1) @ Wp2 + bp2
    gemm_bf16<true, false, true><<<gg, 256, 0, stream>>>(zb, Wp1b, bp1, ((float*)0), p1b, Nn, D, D);
    gemm_bf16<false, true, false><<<gg, 256, 0, stream>>>(p1b, Wp2b, bp2, proj, ((__bf16*)0), Nn, D, D);
}
